// MPNNLayer_39307540692996
// MI455X (gfx1250) — matrix operand order repaired and checked by reading, not compiled, not run on hardware
//
#include <hip/hip_runtime.h>

typedef __attribute__((ext_vector_type(16))) _Float16 v16h;
typedef __attribute__((ext_vector_type(8)))  _Float16 v8h;
typedef __attribute__((ext_vector_type(8)))  float    v8f;
typedef __attribute__((ext_vector_type(4)))  float    v4f;

#define WMMA16(a, b, c) \
  __builtin_amdgcn_wmma_f32_16x16x32_f16(false, (a), false, (b), (short)0, (c), false, false)

// ---------------- constants ----------------
#define NN   10000
#define EE   160000
#define HH   128
#define NIN  128
#define DD   256
#define INV_SCALE (1.0f / 30.0f)
#define EPS  1e-5f

// workspace layout (bytes)
#define OFF_DH    0u
#define OFF_W0S   5120000u           // m_w0 swizzled: 8kt*8nt*512 halves
#define OFF_W1S   5185536u           // m_w1: 4*8*512
#define OFF_DW0S  5218304u           // d_w0: 4*16*512
#define OFF_DW1S  5283840u           // d_w1: 8*16*512
#define OFF_DW2S  5414912u           // d_w2: 8*8*512

__device__ __forceinline__ float gelu_erf(float x) {
  return 0.5f * x * (1.0f + erff(x * 0.70710678118654752f));
}

__device__ __forceinline__ v8f zero8() {
  v8f z;
#pragma unroll
  for (int i = 0; i < 8; ++i) z[i] = 0.0f;
  return z;
}

// A fragment (16x32 f16) from f32 row-major row pointer.
// lane m = lane&15 owns row m; hi = lane>>4.
// halves j0..7 -> k = kbase + j ; j8..15 -> k = kbase + 16 + (j-8), kbase = kt*32 + hi*8
__device__ __forceinline__ v16h load_a_f32(const float* __restrict__ row, int kbase) {
  v4f f0 = *(const v4f*)(row + kbase);
  v4f f1 = *(const v4f*)(row + kbase + 4);
  v4f f2 = *(const v4f*)(row + kbase + 16);
  v4f f3 = *(const v4f*)(row + kbase + 20);
  v16h a;
#pragma unroll
  for (int i = 0; i < 4; ++i) {
    a[i]      = (_Float16)f0[i];
    a[4 + i]  = (_Float16)f1[i];
    a[8 + i]  = (_Float16)f2[i];
    a[12 + i] = (_Float16)f3[i];
  }
  return a;
}

// A fragment from f16 row (LDS staging), same k mapping
__device__ __forceinline__ v16h load_a_h16(const _Float16* row, int kbase) {
  v8h lo = *(const v8h*)(row + kbase);
  v8h hi = *(const v8h*)(row + kbase + 16);
  return __builtin_shufflevector(lo, hi, 0, 1, 2, 3, 4, 5, 6, 7, 8, 9, 10, 11, 12, 13, 14, 15);
}

// B fragment: pre-swizzled, 16 contiguous halves per lane per fragment
__device__ __forceinline__ v16h load_b(const _Float16* __restrict__ buf, int frag, int lane) {
  return *(const v16h*)(buf + ((size_t)((frag << 5) + lane) << 4));
}

// ---------------- weight pre-swizzle ----------------
// W: f32 [KT*32, NT*16] row-major -> dst halves in B-fragment order.
// B layout: lane n = lane&15, hi = lane>>4; half j -> k = kt*32 + hi*8 + (j&7) + (j>>3)*16.
__global__ void swizzle_b_kernel(const float* __restrict__ W, _Float16* __restrict__ dst,
                                 int NT, int total) {
  int idx = blockIdx.x * blockDim.x + threadIdx.x;
  if (idx >= total) return;
  int j    = idx & 15;
  int lane = (idx >> 4) & 31;
  int frag = idx >> 9;
  int kt = frag / NT;
  int nt = frag - kt * NT;
  int k = kt * 32 + (lane >> 4) * 8 + (j & 7) + (j >> 3) * 16;
  int n = nt * 16 + (lane & 15);
  dst[idx] = (_Float16)W[(size_t)k * (NT * 16) + n];
}

// ---------------- edge message kernel ----------------
// one wave (32 threads) per 16-edge tile; EE/16 = 10000 tiles exactly
__global__ void __launch_bounds__(32) edge_kernel(
    const float* __restrict__ hE, const int* __restrict__ eidx,
    const _Float16* __restrict__ w0s, const _Float16* __restrict__ w1s,
    const float* __restrict__ b0, const float* __restrict__ b1,
    float* __restrict__ dh) {
  __shared__ __attribute__((aligned(32))) _Float16 m0s[16 * 128];

  const int lane = threadIdx.x;
  const int m = lane & 15;
  const int hi = lane >> 4;
  const int n = lane & 15;
  const int tile = blockIdx.x;

  const float* row = hE + (size_t)(tile * 16 + m) * (HH + NIN);

  // GEMM1: [16 x 256] x [256 x 128]
  v8f acc[8];
#pragma unroll
  for (int nt = 0; nt < 8; ++nt) acc[nt] = zero8();

#pragma unroll
  for (int kt = 0; kt < 8; ++kt) {
    v16h a = load_a_f32(row, kt * 32 + hi * 8);
#pragma unroll
    for (int nt = 0; nt < 8; ++nt) {
      v16h b = load_b(w0s, kt * 8 + nt, lane);
      acc[nt] = WMMA16(a, b, acc[nt]);
    }
  }

  // bias + GELU -> LDS f16 (C layout: lane owns column n, rows hi*8+r)
#pragma unroll
  for (int nt = 0; nt < 8; ++nt) {
    float bias = b0[nt * 16 + n];
#pragma unroll
    for (int r = 0; r < 8; ++r) {
      m0s[(hi * 8 + r) * 128 + nt * 16 + n] = (_Float16)gelu_erf(acc[nt][r] + bias);
    }
  }
  __syncthreads();

  // GEMM2: [16 x 128] x [128 x 128]
  v8f acc2[8];
#pragma unroll
  for (int nt = 0; nt < 8; ++nt) acc2[nt] = zero8();

#pragma unroll
  for (int kt = 0; kt < 4; ++kt) {
    v16h a = load_a_h16(m0s + m * 128, kt * 32 + hi * 8);
#pragma unroll
    for (int nt = 0; nt < 8; ++nt) {
      v16h b = load_b(w1s, kt * 8 + nt, lane);
      acc2[nt] = WMMA16(a, b, acc2[nt]);
    }
  }

  // bias + GELU + scatter-add into dh[src]
#pragma unroll
  for (int r = 0; r < 8; ++r) {
    int e = tile * 16 + hi * 8 + r;
    int src = eidx[e];  // edge_idx[0][e]
    float* dst = dh + (size_t)src * HH;
#pragma unroll
    for (int nt = 0; nt < 8; ++nt) {
      float v = gelu_erf(acc2[nt][r] + b1[nt * 16 + n]) * INV_SCALE;
      atomicAdd(dst + nt * 16 + n, v);
    }
  }
}

// ---------------- node kernel ----------------
// one wave per 16-node tile; NN/16 = 625 tiles exactly
__global__ void __launch_bounds__(32) node_kernel(
    const float* __restrict__ hV, const float* __restrict__ dh,
    const _Float16* __restrict__ dw0s, const _Float16* __restrict__ dw1s,
    const _Float16* __restrict__ dw2s,
    const float* __restrict__ db0, const float* __restrict__ db1,
    const float* __restrict__ db2,
    const float* __restrict__ g1, const float* __restrict__ be1,
    const float* __restrict__ g2, const float* __restrict__ be2,
    float* __restrict__ out) {
  __shared__ __attribute__((aligned(16))) float    xs[16 * 128];  // LN1 result f32
  __shared__ __attribute__((aligned(32))) _Float16 xh[16 * 128];  // LN1 f16 (A matrix)
  __shared__ __attribute__((aligned(32))) _Float16 t0[16 * 256];
  __shared__ __attribute__((aligned(32))) _Float16 t1[16 * 256];
  __shared__ __attribute__((aligned(16))) float    ys[16 * 128];

  const int lane = threadIdx.x;
  const int m = lane & 15;
  const int hi = lane >> 4;
  const int n = lane & 15;
  const int tile = blockIdx.x;

  // ---- LN1: 2 lanes per row, 64 elements each ----
  const int r2 = lane >> 1;
  const int hlf = lane & 1;
  const size_t gbase = ((size_t)tile * 16 + r2) * HH + hlf * 64;
  const int lbase = r2 * 128 + hlf * 64;

  float s = 0.0f;
#pragma unroll
  for (int i = 0; i < 16; ++i) {
    v4f a = *(const v4f*)(hV + gbase + i * 4);
    v4f b = *(const v4f*)(dh + gbase + i * 4);
    v4f x = a + b;
    *(v4f*)(xs + lbase + i * 4) = x;
    s += x[0] + x[1] + x[2] + x[3];
  }
  s += __shfl_xor(s, 1, 32);
  const float mu = s * (1.0f / 128.0f);
  float vs = 0.0f;
#pragma unroll
  for (int i = 0; i < 64; ++i) {
    float d = xs[lbase + i] - mu;
    vs += d * d;
  }
  vs += __shfl_xor(vs, 1, 32);
  const float rstd = rsqrtf(vs * (1.0f / 128.0f) + EPS);
#pragma unroll
  for (int i = 0; i < 64; ++i) {
    int c = hlf * 64 + i;
    float v = (xs[lbase + i] - mu) * rstd * g1[c] + be1[c];
    xs[lbase + i] = v;
    xh[lbase + i] = (_Float16)v;
  }
  __syncthreads();

  // ---- GEMM1: [16x128] x [128x256] ----
  v8f acc[16];
#pragma unroll
  for (int nt = 0; nt < 16; ++nt) acc[nt] = zero8();
#pragma unroll
  for (int kt = 0; kt < 4; ++kt) {
    v16h a = load_a_h16(xh + m * 128, kt * 32 + hi * 8);
#pragma unroll
    for (int nt = 0; nt < 16; ++nt) {
      v16h b = load_b(dw0s, kt * 16 + nt, lane);
      acc[nt] = WMMA16(a, b, acc[nt]);
    }
  }
#pragma unroll
  for (int nt = 0; nt < 16; ++nt) {
    float bias = db0[nt * 16 + n];
#pragma unroll
    for (int r = 0; r < 8; ++r)
      t0[(hi * 8 + r) * 256 + nt * 16 + n] = (_Float16)gelu_erf(acc[nt][r] + bias);
  }
  __syncthreads();

  // ---- GEMM2: [16x256] x [256x256] ----
#pragma unroll
  for (int nt = 0; nt < 16; ++nt) acc[nt] = zero8();
#pragma unroll
  for (int kt = 0; kt < 8; ++kt) {
    v16h a = load_a_h16(t0 + m * 256, kt * 32 + hi * 8);
#pragma unroll
    for (int nt = 0; nt < 16; ++nt) {
      v16h b = load_b(dw1s, kt * 16 + nt, lane);
      acc[nt] = WMMA16(a, b, acc[nt]);
    }
  }
#pragma unroll
  for (int nt = 0; nt < 16; ++nt) {
    float bias = db1[nt * 16 + n];
#pragma unroll
    for (int r = 0; r < 8; ++r)
      t1[(hi * 8 + r) * 256 + nt * 16 + n] = (_Float16)gelu_erf(acc[nt][r] + bias);
  }
  __syncthreads();

  // ---- GEMM3: [16x256] x [256x128] ----
#pragma unroll
  for (int nt = 0; nt < 8; ++nt) acc[nt] = zero8();
#pragma unroll
  for (int kt = 0; kt < 8; ++kt) {
    v16h a = load_a_h16(t1 + m * 256, kt * 32 + hi * 8);
#pragma unroll
    for (int nt = 0; nt < 8; ++nt) {
      v16h b = load_b(dw2s, kt * 8 + nt, lane);
      acc[nt] = WMMA16(a, b, acc[nt]);
    }
  }
#pragma unroll
  for (int nt = 0; nt < 8; ++nt) {
    float bias = db2[nt * 16 + n];
#pragma unroll
    for (int r = 0; r < 8; ++r)
      ys[(hi * 8 + r) * 128 + nt * 16 + n] = acc[nt][r] + bias;
  }
  __syncthreads();

  // ---- residual + LN2 + store ----
  float s2 = 0.0f;
#pragma unroll
  for (int i = 0; i < 64; ++i) {
    float v = ys[lbase + i] + xs[lbase + i];
    ys[lbase + i] = v;
    s2 += v;
  }
  s2 += __shfl_xor(s2, 1, 32);
  const float mu2 = s2 * (1.0f / 128.0f);
  float vs2 = 0.0f;
#pragma unroll
  for (int i = 0; i < 64; ++i) {
    float d = ys[lbase + i] - mu2;
    vs2 += d * d;
  }
  vs2 += __shfl_xor(vs2, 1, 32);
  const float rstd2 = rsqrtf(vs2 * (1.0f / 128.0f) + EPS);
#pragma unroll
  for (int i = 0; i < 16; ++i) {
    v4f o;
#pragma unroll
    for (int q = 0; q < 4; ++q) {
      int c = hlf * 64 + i * 4 + q;
      o[q] = (ys[lbase + i * 4 + q] - mu2) * rstd2 * g2[c] + be2[c];
    }
    *(v4f*)(out + gbase + i * 4) = o;
  }
}

// ---------------- host launcher ----------------
extern "C" void kernel_launch(void* const* d_in, const int* in_sizes, int n_in,
                              void* d_out, int out_size, void* d_ws, size_t ws_size,
                              hipStream_t stream) {
  const float* hV   = (const float*)d_in[0];
  const float* hE   = (const float*)d_in[1];
  const int*   eidx = (const int*)d_in[2];
  const float* m_w0 = (const float*)d_in[3];
  const float* m_b0 = (const float*)d_in[4];
  const float* m_w1 = (const float*)d_in[5];
  const float* m_b1 = (const float*)d_in[6];
  const float* d_w0 = (const float*)d_in[7];
  const float* d_b0 = (const float*)d_in[8];
  const float* d_w1 = (const float*)d_in[9];
  const float* d_b1 = (const float*)d_in[10];
  const float* d_w2 = (const float*)d_in[11];
  const float* d_b2 = (const float*)d_in[12];
  const float* g1   = (const float*)d_in[13];
  const float* be1  = (const float*)d_in[14];
  const float* g2   = (const float*)d_in[15];
  const float* be2  = (const float*)d_in[16];
  float* out = (float*)d_out;

  char* ws = (char*)d_ws;
  float*    dh   = (float*)(ws + OFF_DH);
  _Float16* w0s  = (_Float16*)(ws + OFF_W0S);
  _Float16* w1s  = (_Float16*)(ws + OFF_W1S);
  _Float16* dw0s = (_Float16*)(ws + OFF_DW0S);
  _Float16* dw1s = (_Float16*)(ws + OFF_DW1S);
  _Float16* dw2s = (_Float16*)(ws + OFF_DW2S);

  // zero the scatter accumulator
  hipMemsetAsync(dh, 0, (size_t)NN * HH * sizeof(float), stream);

  // pre-swizzle weights into WMMA B-fragment order (L2-resident)
  auto swz = [&](const float* W, _Float16* dst, int KT, int NT) {
    int total = KT * NT * 512;
    swizzle_b_kernel<<<(total + 255) / 256, 256, 0, stream>>>(W, dst, NT, total);
  };
  swz(m_w0, w0s, 8, 8);    // 256x128
  swz(m_w1, w1s, 4, 8);    // 128x128
  swz(d_w0, dw0s, 4, 16);  // 128x256
  swz(d_w1, dw1s, 8, 16);  // 256x256
  swz(d_w2, dw2s, 8, 8);   // 256x128

  edge_kernel<<<EE / 16, 32, 0, stream>>>(hE, eidx, w0s, w1s, m_b0, m_b1, dh);
  node_kernel<<<NN / 16, 32, 0, stream>>>(hV, dh, dw0s, dw1s, dw2s,
                                          d_b0, d_b1, d_b2, g1, be1, g2, be2, out);
}
